// ObjectRelationModule_59820304499077
// MI455X (gfx1250) — compile-verified, matrix operand order checked by reading, not run on hardware
//
#include <hip/hip_runtime.h>
#include <hip/hip_bf16.h>

// ---------------------------------------------------------------------------
// ObjectRelationModule for MI455X (gfx1250, wave32, WMMA).
// All GEMMs run through v_wmma_f32_16x16x32_f16 (f16 in, f32 accumulate).
// g = log(max(pe@WgT+bg,1e-6)) materialized fp32 (64MB, L2-resident).
// Attention computes S^T = K*Q^T so the score C-fragment feeds directly into
// the B operand of Y^T = V^T * P^T (no cross-lane transpose). Each attention
// wave handles 32 queries so K/V fragments are loaded once per 32-key tile
// and reused by both query blocks (16 WMMAs per 8 fragment loads).
// ---------------------------------------------------------------------------

typedef __attribute__((ext_vector_type(16))) _Float16 v16h;
typedef __attribute__((ext_vector_type(8)))  _Float16 v8h;
typedef __attribute__((ext_vector_type(8)))  float    v8f;

#define NTOK  1024
#define DFEAT 1024
#define NHEAD 16
#define DHEAD 64

static __device__ __forceinline__ v8f wmma16(v16h a, v16h b, v8f c) {
  // D = A(16x32 f16) * B(32x16 f16) + C(16x16 f32)
  return __builtin_amdgcn_wmma_f32_16x16x32_f16(false, a, false, b,
                                                (short)0, c, false, false);
}

// Load a 16x32 f16 operand tile in CDNA5 A/B VGPR layout from a row-major
// matrix. Per documented layout, lane<16 holds K {k0..k0+7, k0+16..k0+23};
// lane>=16 holds K {k0+8..15, k0+24..31}.
static __device__ __forceinline__ v16h ld_frag_h(const _Float16* __restrict__ base,
                                                 int ld, int row, int k0, int lane) {
  const int hi = lane >> 4, li = lane & 15;
  const _Float16* p = base + (size_t)(row + li) * ld + k0 + hi * 8;
  v8h lo = *(const v8h*)(p);        // 16B load
  v8h hh = *(const v8h*)(p + 16);   // 16B load
  v16h o;
#pragma unroll
  for (int i = 0; i < 8; ++i) { o[i] = lo[i]; o[8 + i] = hh[i]; }
  return o;
}

// Same fragment layout, fp32 source converted in-register.
static __device__ __forceinline__ v16h ld_frag_f32(const float* __restrict__ base,
                                                   int ld, int row, int k0, int lane) {
  const int hi = lane >> 4, li = lane & 15;
  const float* p = base + (size_t)(row + li) * ld + k0 + hi * 8;
  v16h o;
#pragma unroll
  for (int i = 0; i < 8; ++i) { o[i] = (_Float16)p[i]; o[8 + i] = (_Float16)p[16 + i]; }
  return o;
}

// ---------------------------------------------------------------------------
__global__ void k_cvt_f16(const float* __restrict__ in, _Float16* __restrict__ out, int n) {
  int i = blockIdx.x * blockDim.x + threadIdx.x;
  if (i < n) out[i] = (_Float16)in[i];
}

// ---------------------------------------------------------------------------
// Geometry bias: g[h, pair] = log(max(pe[pair,:]·Wg[h,:] + bg[h], 1e-6)),
// pair = i*N + j. One wave = 16 pairs x 16 heads, 2 WMMAs. Streams pe once.
// ---------------------------------------------------------------------------
__global__ void k_geom(const float* __restrict__ pe, const _Float16* __restrict__ wgh,
                       const float* __restrict__ bg, float* __restrict__ gbuf) {
  const int lane = threadIdx.x;
  const int p0 = blockIdx.x * 16;
  v8f c = {0.f, 0.f, 0.f, 0.f, 0.f, 0.f, 0.f, 0.f};
  c = wmma16(ld_frag_f32(pe, 64, p0, 0, lane),  ld_frag_h(wgh, 64, 0, 0, lane),  c);
  c = wmma16(ld_frag_f32(pe, 64, p0, 32, lane), ld_frag_h(wgh, 64, 0, 32, lane), c);

  const int col = lane & 15, hi = lane >> 4;    // col = head h
  const float bh = bg[col];
  float vals[8];
#pragma unroll
  for (int r = 0; r < 8; ++r)
    vals[r] = __logf(fmaxf(c[r] + bh, 1e-6f));  // relu + clamp(1e-6) + log
  float4* dst = (float4*)(gbuf + (size_t)col * (NTOK * NTOK) + p0 + 8 * hi);
  dst[0] = make_float4(vals[0], vals[1], vals[2], vals[3]);
  dst[1] = make_float4(vals[4], vals[5], vals[6], vals[7]);
}

// ---------------------------------------------------------------------------
// Projection GEMM: C = x @ W^T + b. One wave = 16 tokens x 64 out-cols
// (4 column tiles share one A fragment per k-step -> 4 WMMAs per A load).
// vmode 0: store head-major [h][tok][d] (Q, K)
// vmode 1: store transposed [h][d][tok] (V)
// ---------------------------------------------------------------------------
__global__ void k_proj(const _Float16* __restrict__ xh, const _Float16* __restrict__ wh,
                       const float* __restrict__ bias, _Float16* __restrict__ out,
                       int vmode) {
  const int lane = threadIdx.x;
  const int tile = blockIdx.x;          // 64 token-blocks x 16 col-groups
  const int tm = (tile & 63) * 16;
  const int cg = (tile >> 6) * 64;
  v8f c0 = {0.f,0.f,0.f,0.f,0.f,0.f,0.f,0.f};
  v8f c1 = c0, c2 = c0, c3 = c0;
  for (int k0 = 0; k0 < DFEAT; k0 += 32) {
    v16h a = ld_frag_h(xh, DFEAT, tm, k0, lane);
    c0 = wmma16(a, ld_frag_h(wh, DFEAT, cg,      k0, lane), c0);
    c1 = wmma16(a, ld_frag_h(wh, DFEAT, cg + 16, k0, lane), c1);
    c2 = wmma16(a, ld_frag_h(wh, DFEAT, cg + 32, k0, lane), c2);
    c3 = wmma16(a, ld_frag_h(wh, DFEAT, cg + 48, k0, lane), c3);
  }
  const int li = lane & 15, hi = lane >> 4;
  v8f acc[4] = {c0, c1, c2, c3};
#pragma unroll
  for (int t = 0; t < 4; ++t) {
    const int col = cg + 16 * t + li;
    const float bb = bias[col];
    const int h = col >> 6, d = col & 63;
    if (vmode == 0) {
#pragma unroll
      for (int r = 0; r < 8; ++r) {
        int tok = tm + 8 * hi + r;
        out[((size_t)h * NTOK + tok) * DHEAD + d] = (_Float16)(acc[t][r] + bb);
      }
    } else {
      v8h pk;
#pragma unroll
      for (int r = 0; r < 8; ++r) pk[r] = (_Float16)(acc[t][r] + bb);
      *(v8h*)(out + ((size_t)h * DHEAD + d) * NTOK + tm + 8 * hi) = pk;
    }
  }
}

// ---------------------------------------------------------------------------
// One query-block step of fused attention: scores (4 WMMA) + g bias + online
// softmax + PV (4 WMMA). K/V fragments are passed in and shared across blocks.
// ---------------------------------------------------------------------------
static __device__ __forceinline__ void attn_block(
    v16h ak00, v16h ak01, v16h ak10, v16h ak11,
    v16h av0, v16h av1, v16h av2, v16h av3,
    v16h bq0, v16h bq1, const float* __restrict__ grow,
    float& m_run, float& l_run,
    v8f& o0, v8f& o1, v8f& o2, v8f& o3) {
  v8f z = {0.f,0.f,0.f,0.f,0.f,0.f,0.f,0.f};
  v8f s0 = wmma16(ak00, bq0, z);
  s0     = wmma16(ak01, bq1, s0);
  v8f s1 = wmma16(ak10, bq0, z);
  s1     = wmma16(ak11, bq1, s1);
#pragma unroll
  for (int r = 0; r < 8; ++r) { s0[r] += grow[r]; s1[r] += grow[16 + r]; }

  float tmax = -1e30f;
#pragma unroll
  for (int r = 0; r < 8; ++r) tmax = fmaxf(tmax, fmaxf(s0[r], s1[r]));
  tmax = fmaxf(tmax, __shfl_xor(tmax, 16, 32));
  float m_new = fmaxf(m_run, tmax);
  float scale = __expf(m_run - m_new);
  float psum = 0.f;
#pragma unroll
  for (int r = 0; r < 8; ++r) {
    s0[r] = __expf(s0[r] - m_new); psum += s0[r];
    s1[r] = __expf(s1[r] - m_new); psum += s1[r];
  }
  psum += __shfl_xor(psum, 16, 32);
  l_run = l_run * scale + psum;
  m_run = m_new;
#pragma unroll
  for (int r = 0; r < 8; ++r) { o0[r] *= scale; o1[r] *= scale; o2[r] *= scale; o3[r] *= scale; }

  v16h bp;   // P^T C-layout == B-operand layout (no transpose needed)
#pragma unroll
  for (int r = 0; r < 8; ++r) { bp[r] = (_Float16)s0[r]; bp[8 + r] = (_Float16)s1[r]; }
  o0 = wmma16(av0, bp, o0);
  o1 = wmma16(av1, bp, o1);
  o2 = wmma16(av2, bp, o2);
  o3 = wmma16(av3, bp, o3);
}

static __device__ __forceinline__ void attn_store(
    _Float16* __restrict__ Yh, int tok, int h, int hi, float inv,
    v8f o0, v8f o1, v8f o2, v8f o3) {
  _Float16* yp = Yh + (size_t)tok * DFEAT + h * DHEAD + 8 * hi;
  v8h pk;
#pragma unroll
  for (int r = 0; r < 8; ++r) pk[r] = (_Float16)(o0[r] * inv);
  *(v8h*)(yp +  0) = pk;
#pragma unroll
  for (int r = 0; r < 8; ++r) pk[r] = (_Float16)(o1[r] * inv);
  *(v8h*)(yp + 16) = pk;
#pragma unroll
  for (int r = 0; r < 8; ++r) pk[r] = (_Float16)(o2[r] * inv);
  *(v8h*)(yp + 32) = pk;
#pragma unroll
  for (int r = 0; r < 8; ++r) pk[r] = (_Float16)(o3[r] * inv);
  *(v8h*)(yp + 48) = pk;
}

// ---------------------------------------------------------------------------
// Fused attention: one wave per (head, 32 queries). Flash-style online softmax.
// ---------------------------------------------------------------------------
__global__ void k_attn(const _Float16* __restrict__ Qh, const _Float16* __restrict__ Kh,
                       const _Float16* __restrict__ Vt, const float* __restrict__ g,
                       _Float16* __restrict__ Yh) {
  const int h = blockIdx.x;
  const int i0 = blockIdx.y * 32;
  const int lane = threadIdx.x;
  const int hi = lane >> 4, li = lane & 15;

  const _Float16* Qp = Qh + (size_t)h * NTOK * DHEAD;
  const _Float16* Kp = Kh + (size_t)h * NTOK * DHEAD;
  const _Float16* Vp = Vt + (size_t)h * DHEAD * NTOK;
  const float*    gp = g  + (size_t)h * NTOK * NTOK;

  // Q^T B operands for the two query blocks (resident for whole j loop)
  v16h bqA0 = ld_frag_h(Qp, DHEAD, i0,      0,  lane);
  v16h bqA1 = ld_frag_h(Qp, DHEAD, i0,      32, lane);
  v16h bqB0 = ld_frag_h(Qp, DHEAD, i0 + 16, 0,  lane);
  v16h bqB1 = ld_frag_h(Qp, DHEAD, i0 + 16, 32, lane);

  v8f oA0 = {0.f,0.f,0.f,0.f,0.f,0.f,0.f,0.f};
  v8f oA1 = oA0, oA2 = oA0, oA3 = oA0;
  v8f oB0 = oA0, oB1 = oA0, oB2 = oA0, oB3 = oA0;
  float mA = -1e30f, lA = 0.f, mB = -1e30f, lB = 0.f;

  const float* gbaseA = gp + (size_t)(i0 + li) * NTOK + 8 * hi;
  const float* gbaseB = gbaseA + (size_t)16 * NTOK;

  for (int jb = 0; jb < NTOK; jb += 32) {
    // shared K tile (32 keys x 64) and V tile (64 dv x 32 keys) fragments
    v16h ak00 = ld_frag_h(Kp, DHEAD, jb,      0,  lane);
    v16h ak01 = ld_frag_h(Kp, DHEAD, jb,      32, lane);
    v16h ak10 = ld_frag_h(Kp, DHEAD, jb + 16, 0,  lane);
    v16h ak11 = ld_frag_h(Kp, DHEAD, jb + 16, 32, lane);
    v16h av0  = ld_frag_h(Vp, NTOK,  0, jb, lane);
    v16h av1  = ld_frag_h(Vp, NTOK, 16, jb, lane);
    v16h av2  = ld_frag_h(Vp, NTOK, 32, jb, lane);
    v16h av3  = ld_frag_h(Vp, NTOK, 48, jb, lane);

    attn_block(ak00, ak01, ak10, ak11, av0, av1, av2, av3,
               bqA0, bqA1, gbaseA + jb, mA, lA, oA0, oA1, oA2, oA3);
    attn_block(ak00, ak01, ak10, ak11, av0, av1, av2, av3,
               bqB0, bqB1, gbaseB + jb, mB, lB, oB0, oB1, oB2, oB3);
  }

  attn_store(Yh, i0 + li,      h, hi, 1.f / lA, oA0, oA1, oA2, oA3);
  attn_store(Yh, i0 + 16 + li, h, hi, 1.f / lB, oB0, oB1, oB2, oB3);
}

// ---------------------------------------------------------------------------
// Output projection + bias + residual: out = x + Y @ Wy^T + by (fp32 out).
// 16 tokens x 64 cols per wave, A fragment shared across 4 column tiles.
// ---------------------------------------------------------------------------
__global__ void k_yproj(const _Float16* __restrict__ yh, const _Float16* __restrict__ wyh,
                        const float* __restrict__ by, const float* __restrict__ x,
                        float* __restrict__ out) {
  const int lane = threadIdx.x;
  const int tile = blockIdx.x;
  const int tm = (tile & 63) * 16;
  const int cg = (tile >> 6) * 64;
  v8f c0 = {0.f,0.f,0.f,0.f,0.f,0.f,0.f,0.f};
  v8f c1 = c0, c2 = c0, c3 = c0;
  for (int k0 = 0; k0 < DFEAT; k0 += 32) {
    v16h a = ld_frag_h(yh, DFEAT, tm, k0, lane);
    c0 = wmma16(a, ld_frag_h(wyh, DFEAT, cg,      k0, lane), c0);
    c1 = wmma16(a, ld_frag_h(wyh, DFEAT, cg + 16, k0, lane), c1);
    c2 = wmma16(a, ld_frag_h(wyh, DFEAT, cg + 32, k0, lane), c2);
    c3 = wmma16(a, ld_frag_h(wyh, DFEAT, cg + 48, k0, lane), c3);
  }
  const int li = lane & 15, hi = lane >> 4;
  v8f acc[4] = {c0, c1, c2, c3};
#pragma unroll
  for (int t = 0; t < 4; ++t) {
    const int col = cg + 16 * t + li;
    const float bb = by[col];
#pragma unroll
    for (int r = 0; r < 8; ++r) {
      int tok = tm + 8 * hi + r;
      size_t idx = (size_t)tok * DFEAT + col;
      out[idx] = acc[t][r] + bb + x[idx];
    }
  }
}

// ---------------------------------------------------------------------------
extern "C" void kernel_launch(void* const* d_in, const int* in_sizes, int n_in,
                              void* d_out, int out_size, void* d_ws, size_t ws_size,
                              hipStream_t stream) {
  const float* x  = (const float*)d_in[0];
  const float* pe = (const float*)d_in[1];
  const float* Wq = (const float*)d_in[2];
  const float* bq = (const float*)d_in[3];
  const float* Wk = (const float*)d_in[4];
  const float* bk = (const float*)d_in[5];
  const float* Wv = (const float*)d_in[6];
  const float* bv = (const float*)d_in[7];
  const float* Wg = (const float*)d_in[8];
  const float* bg = (const float*)d_in[9];
  const float* Wy = (const float*)d_in[10];
  const float* by = (const float*)d_in[11];
  float* out = (float*)d_out;

  const size_t SZ_M = (size_t)DFEAT * DFEAT;   // 1M elements
  _Float16* xh  = (_Float16*)d_ws;
  _Float16* wqh = xh  + SZ_M;
  _Float16* wkh = wqh + SZ_M;
  _Float16* wvh = wkh + SZ_M;
  _Float16* wyh = wvh + SZ_M;
  _Float16* wgh = wyh + SZ_M;                  // 16*64 = 1024 elems
  _Float16* Qh  = wgh + 1024;                  // [NHEAD][NTOK][64]
  _Float16* Kh  = Qh  + SZ_M;
  _Float16* Vt  = Kh  + SZ_M;                  // [NHEAD][64][NTOK]
  _Float16* Yh  = Vt  + SZ_M;                  // [NTOK][DFEAT]
  float*  gbuf  = (float*)(Yh + SZ_M);         // [NHEAD][NTOK][NTOK] fp32, 64MB

  // fp32 -> f16 staging
  k_cvt_f16<<<(int)(SZ_M / 256), 256, 0, stream>>>(x,  xh,  (int)SZ_M);
  k_cvt_f16<<<(int)(SZ_M / 256), 256, 0, stream>>>(Wq, wqh, (int)SZ_M);
  k_cvt_f16<<<(int)(SZ_M / 256), 256, 0, stream>>>(Wk, wkh, (int)SZ_M);
  k_cvt_f16<<<(int)(SZ_M / 256), 256, 0, stream>>>(Wv, wvh, (int)SZ_M);
  k_cvt_f16<<<(int)(SZ_M / 256), 256, 0, stream>>>(Wy, wyh, (int)SZ_M);
  k_cvt_f16<<<4, 256, 0, stream>>>(Wg, wgh, NHEAD * 64);

  // geometry bias (streams pe once)
  k_geom<<<(NTOK * NTOK) / 16, 32, 0, stream>>>(pe, wgh, bg, gbuf);

  // Q/K head-major, V transposed: 64 token-blocks x 16 col-groups
  k_proj<<<1024, 32, 0, stream>>>(xh, wqh, bq, Qh, 0);
  k_proj<<<1024, 32, 0, stream>>>(xh, wkh, bk, Kh, 0);
  k_proj<<<1024, 32, 0, stream>>>(xh, wvh, bv, Vt, 1);

  // fused flash attention: one wave per (head, 32-query block)
  dim3 ag(NHEAD, NTOK / 32);
  k_attn<<<ag, 32, 0, stream>>>(Qh, Kh, Vt, gbuf, Yh);

  // output projection + residual
  k_yproj<<<1024, 32, 0, stream>>>(Yh, wyh, by, x, out);
}
